// GRUCutsNet_48326972015013
// MI455X (gfx1250) — matrix operand order repaired and checked by reading, not compiled, not run on hardware
//
#include <hip/hip_runtime.h>
#include <hip/hip_bf16.h>
#include <stdint.h>

// Problem dims (from reference)
#define B_  4
#define S_  64
#define N_  1000
#define NP  1024      // padded N for GEMM K / graph rows
#define H_  64
#define E_  8
#define M_  256       // B*H rows of the graph GEMM
#define G3  192       // 3*H

typedef __bf16 bf16_t;
typedef __bf16 v16bf __attribute__((ext_vector_type(16)));
typedef __bf16 v8bf  __attribute__((ext_vector_type(8)));
typedef float  v8f   __attribute__((ext_vector_type(8)));
typedef unsigned int v4u __attribute__((ext_vector_type(4)));
typedef int    v8i   __attribute__((ext_vector_type(8)));
typedef int    v4i   __attribute__((ext_vector_type(4)));

// ---------------------------------------------------------------------------
// P1: row-normalize graphs. fwd/bwd -> bf16 [w][v] padded 1024x1024 (zeros in
// pad). ind -> f32 TRANSPOSED [v][w] (so diffuse_x reads coalesced).
// One wave per output row.
// ---------------------------------------------------------------------------
__global__ void normalize_graphs(const float* __restrict__ fwd,
                                 const float* __restrict__ bwd,
                                 const float* __restrict__ ind,
                                 bf16_t* __restrict__ gf,
                                 bf16_t* __restrict__ gb,
                                 float*  __restrict__ git /* [v][w] */) {
  int wid  = (blockIdx.x * blockDim.x + threadIdx.x) >> 5;
  int lane = threadIdx.x & 31;
  if (wid >= 3 * NP) return;
  int which = wid / NP;
  int row   = wid % NP;
  const float* src = (which == 0) ? fwd : (which == 1) ? bwd : ind;
  float s = 0.f;
  if (row < N_) {
    for (int v = lane; v < N_; v += 32) s += src[row * N_ + v];
    for (int o = 16; o > 0; o >>= 1) s += __shfl_down(s, o, 32);
    s = __shfl(s, 0, 32);
  }
  float inv = (row < N_) ? (1.0f / (s + 1e-6f)) : 0.f;
  for (int v = lane; v < NP; v += 32) {
    float val = (row < N_ && v < N_) ? src[row * N_ + v] * inv : 0.f;
    if (which == 2) {
      git[(size_t)v * NP + row] = val;           // transposed store
    } else {
      bf16_t* dst = (which == 0) ? gf : gb;
      dst[(size_t)row * NP + v] = (bf16_t)val;
    }
  }
}

// Zero the pad columns (n in [1000,1024)) of the bf16 A-layout h buffer once
// per direction; step kernels only ever touch n < 1000.
__global__ void zero_pad_hA(bf16_t* __restrict__ hA) {
  int idx = blockIdx.x * blockDim.x + threadIdx.x;
  const int PADC = NP - N_;
  if (idx >= M_ * PADC) return;
  int m = idx / PADC, c = idx % PADC;
  hA[(size_t)m * NP + N_ + c] = (bf16_t)0.f;
}

// ---------------------------------------------------------------------------
// P2: initialize bf16 A-layout h from h0 (broadcast over batch) and compute
// step-0 x_hat.
// ---------------------------------------------------------------------------
__global__ void init_state(const float* __restrict__ h0,
                           const float* __restrict__ Wi,
                           const float* __restrict__ bi,
                           const float* __restrict__ x,
                           const unsigned char* __restrict__ mask,
                           int s0,
                           bf16_t* __restrict__ hA,
                           float* __restrict__ xhat) {
  int idx = blockIdx.x * blockDim.x + threadIdx.x;
  if (idx >= B_ * N_) return;
  int b = idx / N_, n = idx % N_;
  float dot = 0.f;
#pragma unroll 8
  for (int h = 0; h < H_; ++h) {
    float hv = h0[h * N_ + n];
    hA[(size_t)(b * H_ + h) * NP + n] = (bf16_t)hv;
    dot += Wi[h] * hv;
  }
  float xh1 = dot + bi[0];
  int xi = (b * S_ + s0) * N_ + n;
  xhat[idx] = mask[xi] ? x[xi] : xh1;
}

// ---------------------------------------------------------------------------
// TDM: issue a 2D tile load (bf16 elements) Global -> LDS with 4-dword row
// padding (16 data dwords + 4 pad = 80B LDS pitch). 6-arg builtin form.
// ---------------------------------------------------------------------------
__device__ __forceinline__ void tdm_load_tile_bf16(unsigned lds_off,
                                                   const void* gaddr,
                                                   unsigned tile_d0,   // elems (contig dim)
                                                   unsigned tile_d1,   // rows
                                                   unsigned tensor_d0, // elems
                                                   unsigned stride0) { // elems
  uint64_t ga = (uint64_t)gaddr;
  v4u g0;
  g0[0] = 1u;                                   // count=1 (valid descriptor)
  g0[1] = lds_off;                              // lds_addr
  g0[2] = (unsigned)ga;                         // global_addr[31:0]
  g0[3] = (unsigned)((ga >> 32) & 0x01FFFFFFu)  // global_addr[56:32]
          | (2u << 30);                         // type=2 ("image")
  v8i g1;
  // data_size=1 (2B), pad_enable, pad_interval=3 (16 dwords), pad_amount=3 (4 dwords)
  g1[0] = (int)((1u << 16) | (1u << 20) | (3u << 22) | (3u << 25));
  g1[1] = (int)((tensor_d0 & 0xFFFFu) << 16);               // tensor_dim0 lo16
  g1[2] = (int)((tensor_d0 >> 16) | ((1024u & 0xFFFFu) << 16)); // td0 hi | tensor_dim1 lo16
  g1[3] = (int)((0u) | (tile_d0 << 16));                    // td1 hi16 | tile_dim0
  g1[4] = (int)(tile_d1 & 0xFFFFu);                         // tile_dim1 | tile_dim2=0
  g1[5] = (int)stride0;                                     // tensor_dim0_stride lo32
  g1[6] = 0;                                                // stride0 hi | stride1 lo
  g1[7] = 0;
  v4i z4 = {0, 0, 0, 0};
  v8i z8 = {0, 0, 0, 0, 0, 0, 0, 0};
  __builtin_amdgcn_tensor_load_to_lds(g0, g1, z4, z4, z8, 0);
}

// ---------------------------------------------------------------------------
// K2: h_d[m][w] = sum_v hA[m][v] * Gt[w][v]   (C = A x B, B[v][w] = Gt[w][v])
// bf16 WMMA, f32 accumulate. Macro tile 128x128, 8 waves x (32M x 64W),
// K-loop 32 iters of K=32, TDM double-buffered LDS staging.
// ---------------------------------------------------------------------------
__global__ __launch_bounds__(256)
void graph_wmma(const bf16_t* __restrict__ Gt,  // [NP][NP]
                const bf16_t* __restrict__ hA,  // [M_][NP]
                float* __restrict__ hd) {       // [M_][N_]
  const int bx   = blockIdx.x;          // 16 blocks: 2 (M) x 8 (W)
  const int m0   = (bx & 1) * 128;
  const int w0   = (bx >> 1) * 128;
  const int tid  = threadIdx.x;
  const int wv   = tid >> 5;
  const int lane = tid & 31;
  const int wm   = (wv & 3) * 32;       // wave M offset within tile
  const int wn   = (wv >> 2) * 64;      // wave W offset within tile

  const unsigned TILE = 128u * 80u;     // one staged tile (padded rows)
  __shared__ __align__(16) char smem[2 * 2 * 128 * 80];  // [buf][A|B]
  const unsigned sbase = (unsigned)(size_t)(void*)&smem[0];

  const v8f vzero = {0.f, 0.f, 0.f, 0.f, 0.f, 0.f, 0.f, 0.f};
  v8f acc[2][4];
#pragma unroll
  for (int i = 0; i < 2; ++i)
#pragma unroll
    for (int j = 0; j < 4; ++j) acc[i][j] = vzero;

  const int KITERS = NP / 32;
  const int lhalf = (lane < 16) ? 0 : 1;
  const int lrow  = lane & 15;

  if (wv == 0) {
    tdm_load_tile_bf16(sbase + 0u,    hA + (size_t)m0 * NP, 32, 128, NP, NP);
    tdm_load_tile_bf16(sbase + TILE,  Gt + (size_t)w0 * NP, 32, 128, NP, NP);
  }
  for (int k = 0; k < KITERS; ++k) {
    const int buf = k & 1;
    if (wv == 0) {
      if (k + 1 < KITERS) {
        unsigned nb = sbase + (unsigned)((buf ^ 1) * 2) * TILE;
        tdm_load_tile_bf16(nb,        hA + (size_t)m0 * NP + (k + 1) * 32, 32, 128, NP, NP);
        tdm_load_tile_bf16(nb + TILE, Gt + (size_t)w0 * NP + (k + 1) * 32, 32, 128, NP, NP);
        __builtin_amdgcn_s_wait_tensorcnt(2);   // current buffer's 2 loads done
      } else {
        __builtin_amdgcn_s_wait_tensorcnt(0);
      }
    }
    __syncthreads();

    const char* Abase = smem + buf * 2 * TILE;
    const char* Bbase = Abase + TILE;

    // A fragments: lanes 0-15 hold K {0..7,16..23}, lanes 16-31 K {8..15,24..31}
    v16bf afrag[2];
#pragma unroll
    for (int i = 0; i < 2; ++i) {
      const char* p = Abase + (wm + i * 16 + lrow) * 80 + lhalf * 16;
      v8bf lo = *(const v8bf*)(const void*)p;
      v8bf hi = *(const v8bf*)(const void*)(p + 32);
      afrag[i] = __builtin_shufflevector(lo, hi, 0, 1, 2, 3, 4, 5, 6, 7,
                                         8, 9, 10, 11, 12, 13, 14, 15);
    }
    // B fragments: lanes 0-15 hold K {0..7,16..23}, lanes 16-31 K {8..15,24..31}
    v16bf bfrag[4];
#pragma unroll
    for (int j = 0; j < 4; ++j) {
      const char* p = Bbase + (wn + j * 16 + lrow) * 80 + lhalf * 16;
      v8bf lo = *(const v8bf*)(const void*)p;
      v8bf hi = *(const v8bf*)(const void*)(p + 32);
      bfrag[j] = __builtin_shufflevector(lo, hi, 0, 1, 2, 3, 4, 5, 6, 7,
                                         8, 9, 10, 11, 12, 13, 14, 15);
    }
#pragma unroll
    for (int i = 0; i < 2; ++i)
#pragma unroll
      for (int j = 0; j < 4; ++j)
        acc[i][j] = __builtin_amdgcn_wmma_f32_16x16x32_bf16(
            false, afrag[i], false, bfrag[j], (short)0, acc[i][j], false, false);
    __syncthreads();
  }

  // Epilogue: C layout — VGPR r: lanes 0-15 -> M=r, lanes 16-31 -> M=8+r
  const int rbase = m0 + wm + ((lane < 16) ? 0 : 8);
  const int cbase = w0 + wn + lrow;
#pragma unroll
  for (int i = 0; i < 2; ++i) {
#pragma unroll
    for (int j = 0; j < 4; ++j) {
      int col = cbase + j * 16;
      if (col < N_) {
        int rb = rbase + i * 16;
#pragma unroll
        for (int r = 0; r < 8; ++r)
          hd[(size_t)(rb + r) * N_ + col] = acc[i][j][r];
      }
    }
  }
}

// ---------------------------------------------------------------------------
// K2b: xw[b][w] = sum_v xhat[b][v] * IG[w][v]  (IG stored transposed [v][w])
// ---------------------------------------------------------------------------
__global__ void diffuse_x(const float* __restrict__ IGt,  // [v][w] padded NP
                          const float* __restrict__ xhat, // [B][N]
                          float* __restrict__ xw) {       // [B][N]
  int idx = blockIdx.x * blockDim.x + threadIdx.x;
  if (idx >= B_ * N_) return;
  int b = idx / N_, w = idx % N_;
  const float* xb = xhat + b * N_;
  float s = 0.f;
  for (int v = 0; v < N_; ++v) s += IGt[(size_t)v * NP + w] * xb[v];
  xw[idx] = s;
}

// ---------------------------------------------------------------------------
// K3: GRU cell per (b,n): gh = Whh @ h_d column (Whh in LDS), gates, h_new.
// Writes bf16 A-layout h for the next WMMA step, next-step x_hat inline,
// and either f_repr (fwd) or the fused output MLP (bwd).
// ---------------------------------------------------------------------------
__global__ __launch_bounds__(256)
void gru_step(const float* __restrict__ hd,    // [M_][N_]
              const float* __restrict__ xw,    // [B][N]
              const float* __restrict__ Whh,   // [192][64]
              const float* __restrict__ Wih,   // [192]
              const float* __restrict__ bih,   // [192]
              const float* __restrict__ bhh,   // [192]
              const float* __restrict__ Wi,    // [64]
              const float* __restrict__ bi,    // [1]
              const float* __restrict__ x,     // [B][S][N]
              const unsigned char* __restrict__ mask,
              bf16_t* __restrict__ hA,
              float* __restrict__ xhat_next,
              int s_next,                       // <0: no next step
              float* __restrict__ frepr_slice,  // fwd: &frepr[s][0][0], else null
              const float* __restrict__ frepr_read, // bwd: &frepr[s][0][0]
              const float* __restrict__ emb,   // [E][N]
              const float* __restrict__ Wm1,   // [64][137]
              const float* __restrict__ bm1,   // [64]
              const float* __restrict__ Wm2,   // [64]
              const float* __restrict__ bm2,   // [1]
              float* __restrict__ out,          // bwd: full output, else null
              int s_out) {
  __shared__ float sWhh[G3 * H_];
  __shared__ float sBih[G3], sBhh[G3], sWih[G3];
  __shared__ float sWi[H_];
  for (int i = threadIdx.x; i < G3 * H_; i += blockDim.x) sWhh[i] = Whh[i];
  for (int i = threadIdx.x; i < G3; i += blockDim.x) {
    sBih[i] = bih[i]; sBhh[i] = bhh[i]; sWih[i] = Wih[i];
  }
  if (threadIdx.x < H_) sWi[threadIdx.x] = Wi[threadIdx.x];
  __syncthreads();

  int idx = blockIdx.x * blockDim.x + threadIdx.x;
  if (idx >= B_ * N_) return;
  int b = idx / N_, n = idx % N_;

  float hdv[H_];
#pragma unroll 8
  for (int h = 0; h < H_; ++h) hdv[h] = hd[(size_t)(b * H_ + h) * N_ + n];

  const float xv = xw[idx];
  float hnew[H_];
  float xdot = 0.f;
  for (int h = 0; h < H_; ++h) {
    float gr = sBhh[h], gz = sBhh[64 + h], gg = sBhh[128 + h];
#pragma unroll 8
    for (int k = 0; k < H_; ++k) {
      float hk = hdv[k];
      gr += sWhh[h * H_ + k] * hk;
      gz += sWhh[(64 + h) * H_ + k] * hk;
      gg += sWhh[(128 + h) * H_ + k] * hk;
    }
    float ir = sWih[h] * xv + sBih[h];
    float iz = sWih[64 + h] * xv + sBih[64 + h];
    float ig = sWih[128 + h] * xv + sBih[128 + h];
    float r = 1.f / (1.f + __expf(-(ir + gr)));
    float z = 1.f / (1.f + __expf(-(iz + gz)));
    float g = tanhf(ig + r * gg);
    float hv = (1.f - z) * g + z * hdv[h];
    hnew[h] = hv;
    xdot += sWi[h] * hv;
    hA[(size_t)(b * H_ + h) * NP + n] = (bf16_t)hv;
    if (frepr_slice) frepr_slice[(size_t)(b * H_ + h) * N_ + n] = hv;
  }

  if (s_next >= 0) {
    float xh1 = xdot + bi[0];
    int xi = (b * S_ + s_next) * N_ + n;
    xhat_next[idx] = mask[xi] ? x[xi] : xh1;
  }

  if (out) {  // fused output MLP (backward pass): feat=[f_repr,h_new,m,emb]
    int mi = (b * S_ + s_out) * N_ + n;
    float mval = mask[mi] ? 1.f : 0.f;
    float hm[H_];
#pragma unroll 8
    for (int o = 0; o < H_; ++o)
      hm[o] = bm1[o] + Wm1[o * 137 + 128] * mval;
    for (int h = 0; h < H_; ++h) {
      float frh = frepr_read[(size_t)(b * H_ + h) * N_ + n];
      float hnh = hnew[h];
#pragma unroll 8
      for (int o = 0; o < H_; ++o)
        hm[o] += Wm1[o * 137 + h] * frh + Wm1[o * 137 + 64 + h] * hnh;
    }
#pragma unroll
    for (int e = 0; e < E_; ++e) {
      float ev = emb[e * N_ + n];
#pragma unroll 8
      for (int o = 0; o < H_; ++o) hm[o] += Wm1[o * 137 + 129 + e] * ev;
    }
    float oacc = bm2[0];
#pragma unroll 8
    for (int o = 0; o < H_; ++o) oacc += Wm2[o] * fmaxf(hm[o], 0.f);
    out[(size_t)(b * S_ + s_out) * N_ + n] = oacc;
  }
}

// ---------------------------------------------------------------------------
extern "C" void kernel_launch(void* const* d_in, const int* in_sizes, int n_in,
                              void* d_out, int out_size, void* d_ws, size_t ws_size,
                              hipStream_t stream) {
  (void)in_sizes; (void)n_in; (void)out_size; (void)ws_size;
  const float* x          = (const float*)d_in[0];
  const unsigned char* mk = (const unsigned char*)d_in[1];  // bool -> 1 byte
  const float* fwd_graph  = (const float*)d_in[2];
  const float* bwd_graph  = (const float*)d_in[3];
  const float* ind_graph  = (const float*)d_in[4];
  const float* emb        = (const float*)d_in[5];
  const float* h0_[2]  = {(const float*)d_in[6],  (const float*)d_in[15]};
  const float* Wi_[2]  = {(const float*)d_in[7],  (const float*)d_in[16]};
  const float* bi_[2]  = {(const float*)d_in[8],  (const float*)d_in[17]};
  const float* Wih_[2] = {(const float*)d_in[11], (const float*)d_in[20]};
  const float* Whh_[2] = {(const float*)d_in[12], (const float*)d_in[21]};
  const float* bih_[2] = {(const float*)d_in[13], (const float*)d_in[22]};
  const float* bhh_[2] = {(const float*)d_in[14], (const float*)d_in[23]};
  const float* Wm1 = (const float*)d_in[24];
  const float* bm1 = (const float*)d_in[25];
  const float* Wm2 = (const float*)d_in[26];
  const float* bm2 = (const float*)d_in[27];
  float* out = (float*)d_out;

  char* ws = (char*)d_ws;
  size_t off = 0;
  auto alloc = [&](size_t bytes) -> void* {
    void* p = ws + off;
    off = (off + bytes + 255) & ~(size_t)255;
    return p;
  };
  bf16_t* Gf   = (bf16_t*)alloc((size_t)NP * NP * 2);
  bf16_t* Gb   = (bf16_t*)alloc((size_t)NP * NP * 2);
  float*  IGt  = (float*) alloc((size_t)NP * NP * 4);
  bf16_t* hA   = (bf16_t*)alloc((size_t)M_ * NP * 2);
  float*  hd   = (float*) alloc((size_t)M_ * N_ * 4);
  float*  xhat = (float*) alloc((size_t)B_ * N_ * 4);
  float*  xw   = (float*) alloc((size_t)B_ * N_ * 4);
  float*  frepr= (float*) alloc((size_t)S_ * M_ * N_ * 4);  // ~65.5 MB

  // P1: normalize graphs (one wave per row; 3*1024 rows)
  normalize_graphs<<<(3 * NP * 32 + 255) / 256, 256, 0, stream>>>(
      fwd_graph, bwd_graph, ind_graph, Gf, Gb, IGt);

  const int grid_bn = (B_ * N_ + 255) / 256;  // 16
  for (int dir = 0; dir < 2; ++dir) {
    const bf16_t* G = dir ? Gb : Gf;
    zero_pad_hA<<<(M_ * (NP - N_) + 255) / 256, 256, 0, stream>>>(hA);
    int s0 = dir ? (S_ - 1) : 0;
    init_state<<<grid_bn, 256, 0, stream>>>(h0_[dir], Wi_[dir], bi_[dir],
                                            x, mk, s0, hA, xhat);
    for (int t = 0; t < S_; ++t) {
      graph_wmma<<<16, 256, 0, stream>>>(G, hA, hd);
      diffuse_x<<<grid_bn, 256, 0, stream>>>(IGt, xhat, xw);
      int s_cur  = dir ? (S_ - 1 - t) : t;
      int s_next = (t + 1 < S_) ? (dir ? (S_ - 2 - t) : (t + 1)) : -1;
      if (dir == 0) {
        gru_step<<<grid_bn, 256, 0, stream>>>(
            hd, xw, Whh_[0], Wih_[0], bih_[0], bhh_[0], Wi_[0], bi_[0],
            x, mk, hA, xhat, s_next,
            frepr + (size_t)s_cur * M_ * N_,            // store f_repr slice
            nullptr, nullptr, nullptr, nullptr, nullptr, nullptr,
            nullptr, 0);
      } else {
        gru_step<<<grid_bn, 256, 0, stream>>>(
            hd, xw, Whh_[1], Wih_[1], bih_[1], bhh_[1], Wi_[1], bi_[1],
            x, mk, hA, xhat, s_next,
            nullptr,
            frepr + (size_t)s_cur * M_ * N_,            // read f_repr slice
            emb, Wm1, bm1, Wm2, bm2,
            out, s_cur);                                 // fused MLP output
      }
    }
  }
}
